// TS4_42099269436313
// MI455X (gfx1250) — matrix operand order repaired and checked by reading, not compiled, not run on hardware
//
#include <hip/hip_runtime.h>
#include <hip/hip_bf16.h>
#include <math.h>

typedef __attribute__((ext_vector_type(16))) __bf16 bf16x16;
typedef __attribute__((ext_vector_type(8)))  float  f32x8;
typedef unsigned int u32;

__device__ __forceinline__ float gelu_f(float x) {
  return 0.5f * x * (1.f + erff(x * 0.70710678118654752440f));
}
__device__ __forceinline__ float silu_f(float x) {
  return x / (1.f + __expf(-x));
}
__device__ __forceinline__ f32x8 fzero8() {
  f32x8 z;
#pragma unroll
  for (int i = 0; i < 8; ++i) z[i] = 0.f;
  return z;
}

// ---- WMMA fragment loaders (bf16 16x16x32, layouts per CDNA5 ISA) ----
// A: 16x32, rows m = lane&15 ; lanes 0-15 K=0..7,16..23 ; lanes 16-31 K=8..15,24..31
__device__ __forceinline__ bf16x16 load_afrag(const __bf16* A, int lda, int lane) {
  bf16x16 a;
  u32* au = (u32*)&a;
  const __bf16* base = A + (lane & 15) * lda + ((lane >> 4) << 3);
#pragma unroll
  for (int v = 0; v < 8; ++v) {
    int k0 = ((v & 4) << 2) + ((v & 3) << 1);   // 0,2,4,6,16,18,20,22
    au[v] = *(const u32*)(base + k0);
  }
  return a;
}
// B: 32x16 stored transposed in LDS as [n][k]; lanes 0-15 K=0..7,16..23 ; lanes 16-31 K=8..15,24..31
__device__ __forceinline__ bf16x16 load_bfrag(const __bf16* W, int ldw, int lane) {
  bf16x16 b;
  u32* bu = (u32*)&b;
  const __bf16* base = W + (lane & 15) * ldw + ((lane >> 4) << 3);
#pragma unroll
  for (int v = 0; v < 8; ++v) bu[v] = *(const u32*)(base + ((v & 4) << 2) + ((v & 3) << 1));
  return b;
}
__device__ __forceinline__ f32x8 wmma_bf16(bf16x16 a, bf16x16 b, f32x8 c) {
  return __builtin_amdgcn_wmma_f32_16x16x32_bf16(false, a, false, b, (short)0, c,
                                                 false, false);
}
// prefetch one 64B span of the next weight K-chunk (emits global_prefetch_b8)
__device__ __forceinline__ void prefetch_wchunk(const float* w, int kb_next,
                                                int ncols, int tid) {
  const float* a = w + (kb_next + (tid >> 3)) * ncols + ((tid & 7) << 4);
  __builtin_prefetch(a, 0, 1);
}

// =====================  time embedding: c = TE(t) + TE(dt)  ====================
__global__ void k_timeemb(const float* t_val, const float* dt_val,
                          const float* w1t, const float* b1t,
                          const float* w2t, const float* b2t,
                          const float* w1d, const float* b1d,
                          const float* w2d, const float* b2d, float* c) {
  __shared__ float se[256];
  __shared__ float sh[128];
  int tid = threadIdx.x;
  float cacc = 0.f;
  for (int e = 0; e < 2; ++e) {
    float tv = (e == 0) ? t_val[0] : dt_val[0];
    const float* w1 = e ? w1d : w1t; const float* b1 = e ? b1d : b1t;
    const float* w2 = e ? w2d : w2t; const float* b2 = e ? b2d : b2t;
    int i = tid & 127;
    float arg = tv * __expf(-9.210340371976184f * (float)i / 128.f);
    se[tid] = (tid < 128) ? __cosf(arg) : __sinf(arg);
    __syncthreads();
    if (tid < 128) {
      float a = b1[tid];
      for (int j = 0; j < 256; ++j) a += se[j] * w1[j * 128 + tid];
      sh[tid] = silu_f(a);
    }
    __syncthreads();
    if (tid < 128) {
      float a = b2[tid];
      for (int j = 0; j < 128; ++j) a += sh[j] * w2[j * 128 + tid];
      cacc += a;
    }
    __syncthreads();
  }
  if (tid < 128) c[tid] = cacc;
}

// =====================  per-layer xq = silu(c) @ wt_q + b  =====================
__global__ void k_xq(const float* c, const float* wq, const float* bq, float* xq) {
  __shared__ float sc[128];
  int tid = threadIdx.x;
  if (tid < 128) sc[tid] = silu_f(c[tid]);
  __syncthreads();
  int o = blockIdx.x * 256 + tid;          // 16 blocks x 256 = 4096 outputs
  float a = bq[o];
  for (int j = 0; j < 128; ++j) a += sc[j] * wq[j * 4096 + o];
  xq[o] = a;
}

// ==========  x embedding: h = fc2( gelu( fc1(x) ) ), fc2 via WMMA  ============
__global__ void k_x_emb(const float* x, const float* w1, const float* b1,
                        const float* w2, const float* b2, float* h, int n_tok) {
  __shared__ float xs[128][8];
  __shared__ float sb2[128];
  __shared__ __align__(16) __bf16 sA[128 * 36];
  __shared__ __align__(16) __bf16 sW[128 * 36];
  int tid = threadIdx.x, lane = tid & 31, wid = tid >> 5;
  int row0 = blockIdx.x * 128;
  for (int i = tid; i < 128 * 8; i += 256) {
    int r = i >> 3, j = i & 7, n = row0 + r;
    xs[r][j] = (n < n_tok) ? x[n * 8 + j] : 0.f;
  }
  if (tid < 128) sb2[tid] = b2[tid];
  __syncthreads();
  f32x8 acc[8];
#pragma unroll
  for (int t = 0; t < 8; ++t) acc[t] = fzero8();
  for (int kc = 0; kc < 8; ++kc) {            // K = 256, chunks of 32
    int kb = kc * 32;
    for (int i = tid; i < 128 * 32; i += 256) {
      int r = i >> 5, kk = i & 31, cidx = kb + kk;
      float a = b1[cidx];
#pragma unroll
      for (int j = 0; j < 8; ++j) a += xs[r][j] * w1[j * 256 + cidx];
      sA[r * 36 + kk] = (__bf16)gelu_f(a);
    }
    for (int i = tid; i < 128 * 32; i += 256) {
      int kk = i >> 7, col = i & 127;
      sW[col * 36 + kk] = (__bf16)w2[(kb + kk) * 128 + col];
    }
    if (kc < 7) prefetch_wchunk(w2, kb + 32, 128, tid);
    __syncthreads();
    bf16x16 a = load_afrag(sA + (wid << 4) * 36, 36, lane);
#pragma unroll
    for (int t = 0; t < 8; ++t)
      acc[t] = wmma_bf16(a, load_bfrag(sW + (t << 4) * 36, 36, lane), acc[t]);
    __syncthreads();
  }
  int hi = lane >> 4, nn = lane & 15;
#pragma unroll
  for (int t = 0; t < 8; ++t) {
    int col = (t << 4) + nn;
#pragma unroll
    for (int r = 0; r < 8; ++r) {
      int row = row0 + (wid << 4) + (hi << 3) + r;
      if (row < n_tok) h[row * 128 + col] = acc[t][r] + sb2[col];
    }
  }
}

// ======  kv = LN1(h) @ wt_kv + b  (WMMA, bf16 output; 256 cols = 16 tiles) =====
__global__ void k_kv(const float* h, const float* g, const float* be,
                     const float* wkv, const float* bkv, __bf16* kv, int n_tok) {
  __shared__ float smu[128], srs[128], sg[128], sbe[128], sbk[256];
  __shared__ __align__(16) __bf16 sA[128 * 36];
  __shared__ __align__(16) __bf16 sW[256 * 36];
  int tid = threadIdx.x, lane = tid & 31, wid = tid >> 5;
  int row0 = blockIdx.x * 128;
  if (tid < 128) { sg[tid] = g[tid]; sbe[tid] = be[tid]; }
  sbk[tid] = bkv[tid]; // 256 threads == 256 biases
  if (tid < 128) {
    int n = row0 + tid;
    float s = 0.f, ss = 0.f;
    if (n < n_tok)
      for (int j = 0; j < 128; ++j) { float v = h[n * 128 + j]; s += v; ss += v * v; }
    float mu = s * (1.f / 128.f);
    smu[tid] = mu;
    srs[tid] = rsqrtf(ss * (1.f / 128.f) - mu * mu + 1e-5f);
  }
  __syncthreads();
  f32x8 acc[16];
#pragma unroll
  for (int t = 0; t < 16; ++t) acc[t] = fzero8();
  for (int kc = 0; kc < 4; ++kc) {
    int kb = kc * 32;
    for (int i = tid; i < 128 * 32; i += 256) {
      int r = i >> 5, kk = i & 31, cc = kb + kk, n = row0 + r;
      float v = (n < n_tok) ? (h[n * 128 + cc] - smu[r]) * srs[r] * sg[cc] + sbe[cc] : 0.f;
      sA[r * 36 + kk] = (__bf16)v;
    }
    for (int i = tid; i < 256 * 32; i += 256) {
      int kk = i >> 8, col = i & 255;
      sW[col * 36 + kk] = (__bf16)wkv[(kb + kk) * 256 + col];
    }
    if (kc < 3) prefetch_wchunk(wkv, kb + 32, 256, tid);
    __syncthreads();
    bf16x16 a = load_afrag(sA + (wid << 4) * 36, 36, lane);
#pragma unroll
    for (int t = 0; t < 16; ++t)
      acc[t] = wmma_bf16(a, load_bfrag(sW + (t << 4) * 36, 36, lane), acc[t]);
    __syncthreads();
  }
  int hi = lane >> 4, nn = lane & 15;
#pragma unroll
  for (int t = 0; t < 16; ++t) {
    int col = (t << 4) + nn;
#pragma unroll
    for (int r = 0; r < 8; ++r) {
      int row = row0 + (wid << 4) + (hi << 3) + r;
      if (row < n_tok) kv[row * 256 + col] = (__bf16)(acc[t][r] + sbk[col]);
    }
  }
}

// ===  slice pass1: per-token softmax over 32 slices, block-partial sum w (x) v ===
__global__ void k_slice1(const __bf16* kv, const float* xq, const float* alpha,
                         float* gpart, int n_tok) {
  __shared__ float sxq[8 * 32 * 16];
  __shared__ __align__(16) __bf16 swt[256 * 33];
  __shared__ __align__(16) __bf16 svt[256 * 17];
  int tid = threadIdx.x;
  for (int i = tid; i < 4096; i += 256) sxq[i] = xq[i];
  __syncthreads();
  int n = blockIdx.x * 256 + tid;
  bool ok = n < n_tok;
  for (int hh = 0; hh < 8; ++hh) {
    float kvec[16];
#pragma unroll
    for (int d = 0; d < 16; ++d)
      kvec[d] = ok ? (float)kv[n * 256 + hh * 16 + d] : 0.f;
    float al = alpha[hh];
    float w[32], mx = -1e30f;
#pragma unroll
    for (int m = 0; m < 32; ++m) {
      const float* q = &sxq[(hh * 32 + m) * 16];
      float s = 0.f;
#pragma unroll
      for (int d = 0; d < 16; ++d) s += q[d] * kvec[d];
      s *= al; w[m] = s; mx = fmaxf(mx, s);
    }
    float sum = 0.f;
#pragma unroll
    for (int m = 0; m < 32; ++m) { w[m] = __expf(w[m] - mx); sum += w[m]; }
    float inv = 1.f / sum;
#pragma unroll
    for (int m = 0; m < 32; ++m) swt[tid * 33 + m] = (__bf16)(ok ? w[m] * inv : 0.f);
#pragma unroll
    for (int d = 0; d < 16; ++d)
      svt[tid * 17 + d] = ok ? kv[n * 256 + 128 + hh * 16 + d] : (__bf16)0.f;
    svt[tid * 17 + 16] = (__bf16)1.f;     // column 16 accumulates slice_norm
    __syncthreads();
    for (int o = tid; o < 544; o += 256) {  // 32 slices x 17 (16 dims + norm)
      int m = o / 17, d = o - m * 17;
      float s = 0.f;
      for (int t2 = 0; t2 < 256; ++t2)
        s += (float)swt[t2 * 33 + m] * (float)svt[t2 * 17 + d];
      __builtin_nontemporal_store(s, &gpart[blockIdx.x * 4352 + hh * 544 + o]);
    }
    __syncthreads();
  }
}

// deterministic cross-block reduction of the 4352 slice partials
__global__ void k_reduce(const float* part, float* gacc, int nblocks) {
  int o = blockIdx.x * 256 + threadIdx.x;
  if (o >= 4352) return;
  float s = 0.f;
  for (int b = 0; b < nblocks; ++b)
    s += __builtin_nontemporal_load(&part[(size_t)b * 4352 + o]);
  gacc[o] = s;
}

// ====  slice mix: normalize token, qkv projection, 32x32 attention per head ====
__global__ void k_slice_mix(const float* gacc, const float* qkvw, float* ot) {
  __shared__ float st[512], sq[512], sk[512], sv[512], sat[1024];
  int tid = threadIdx.x;
  for (int hh = 0; hh < 8; ++hh) {
    for (int i = tid; i < 512; i += 256) {
      int m = i >> 4, d = i & 15;
      st[i] = gacc[hh * 544 + m * 17 + d] / (gacc[hh * 544 + m * 17 + 16] + 1e-5f);
    }
    __syncthreads();
    for (int i = tid; i < 32 * 48; i += 256) {
      int m = i / 48, e = i - m * 48;
      float s = 0.f;
      for (int d = 0; d < 16; ++d) s += st[m * 16 + d] * qkvw[(hh * 16 + d) * 48 + e];
      if (e < 16) sq[m * 16 + e] = s;
      else if (e < 32) sk[m * 16 + (e - 16)] = s;
      else sv[m * 16 + (e - 32)] = s;
    }
    __syncthreads();
    for (int i = tid; i < 1024; i += 256) {
      int qi = i >> 5, ki = i & 31;
      float s = 0.f;
      for (int d = 0; d < 16; ++d) s += sq[qi * 16 + d] * sk[ki * 16 + d];
      sat[i] = s * 0.25f;                         // HD^-0.5
    }
    __syncthreads();
    if (tid < 32) {
      float mx = -1e30f;
      for (int kq = 0; kq < 32; ++kq) mx = fmaxf(mx, sat[tid * 32 + kq]);
      float sum = 0.f;
      for (int kq = 0; kq < 32; ++kq) {
        float e2 = __expf(sat[tid * 32 + kq] - mx);
        sat[tid * 32 + kq] = e2; sum += e2;
      }
      float inv = 1.f / sum;
      for (int kq = 0; kq < 32; ++kq) sat[tid * 32 + kq] *= inv;
    }
    __syncthreads();
    for (int i = tid; i < 512; i += 256) {
      int qi = i >> 4, d = i & 15;
      float s = 0.f;
      for (int kq = 0; kq < 32; ++kq) s += sat[qi * 32 + kq] * sv[kq * 16 + d];
      ot[(hh * 32 + qi) * 16 + d] = s;
    }
    __syncthreads();
  }
}

// ====  apply: rebuild slice weights, scatter out_token, out-proj WMMA + resid ===
__global__ void k_attn_out(float* h, const __bf16* kv, const float* xq,
                           const float* alpha, const float* ot, const float* wo,
                           const float* bo, int n_tok) {
  __shared__ float sxq[8 * 32 * 16];
  __shared__ float sbo[128];
  __shared__ __align__(16) __bf16 sA[128 * 132];
  __shared__ __align__(16) __bf16 sW[128 * 36];
  int tid = threadIdx.x, lane = tid & 31, wid = tid >> 5;
  int row0 = blockIdx.x * 128;
  for (int i = tid; i < 4096; i += 256) sxq[i] = xq[i];
  if (tid < 128) sbo[tid] = bo[tid];
  __syncthreads();
  // phase 1: each (token, half) pair handles 4 heads
  int tok = tid >> 1, half = tid & 1;
  int n = row0 + tok;
  bool ok = n < n_tok;
  for (int hh = half * 4; hh < half * 4 + 4; ++hh) {
    float kvec[16];
#pragma unroll
    for (int d = 0; d < 16; ++d)
      kvec[d] = ok ? (float)kv[n * 256 + hh * 16 + d] : 0.f;
    float al = alpha[hh];
    float w[32], mx = -1e30f;
#pragma unroll
    for (int m = 0; m < 32; ++m) {
      const float* q = &sxq[(hh * 32 + m) * 16];
      float s = 0.f;
#pragma unroll
      for (int d = 0; d < 16; ++d) s += q[d] * kvec[d];
      s *= al; w[m] = s; mx = fmaxf(mx, s);
    }
    float sum = 0.f;
#pragma unroll
    for (int m = 0; m < 32; ++m) { w[m] = __expf(w[m] - mx); sum += w[m]; }
    float inv = 1.f / sum;
    for (int d = 0; d < 16; ++d) {
      float s = 0.f;
#pragma unroll
      for (int m = 0; m < 32; ++m) s += ot[(hh * 32 + m) * 16 + d] * w[m];
      sA[tok * 132 + hh * 16 + d] = (__bf16)(ok ? s * inv : 0.f);
    }
  }
  __syncthreads();
  // phase 2: out = A @ wo + bo ; h += out
  f32x8 acc[8];
#pragma unroll
  for (int t = 0; t < 8; ++t) acc[t] = fzero8();
  for (int kc = 0; kc < 4; ++kc) {
    int kb = kc * 32;
    for (int i = tid; i < 128 * 32; i += 256) {
      int kk = i >> 7, col = i & 127;
      sW[col * 36 + kk] = (__bf16)wo[(kb + kk) * 128 + col];
    }
    if (kc < 3) prefetch_wchunk(wo, kb + 32, 128, tid);
    __syncthreads();
    bf16x16 a = load_afrag(sA + (wid << 4) * 132 + kb, 132, lane);
#pragma unroll
    for (int t = 0; t < 8; ++t)
      acc[t] = wmma_bf16(a, load_bfrag(sW + (t << 4) * 36, 36, lane), acc[t]);
    __syncthreads();
  }
  int hi = lane >> 4, nn = lane & 15;
#pragma unroll
  for (int t = 0; t < 8; ++t) {
    int col = (t << 4) + nn;
#pragma unroll
    for (int r = 0; r < 8; ++r) {
      int row = row0 + (wid << 4) + (hi << 3) + r;
      if (row < n_tok) h[row * 128 + col] += acc[t][r] + sbo[col];
    }
  }
}

// =======  MLP: h += fc2( gelu( fc1( LN2(h) ) ) ), both GEMMs via WMMA  ========
__global__ void k_mlp(float* h, const float* g, const float* be, const float* w1,
                      const float* b1, const float* w2, const float* b2, int n_tok) {
  __shared__ float smu[128], srs[128], sg[128], sbe[128], sb1[128], sb2[128];
  __shared__ __align__(16) __bf16 sA[128 * 36];
  __shared__ __align__(16) __bf16 sW[128 * 36];
  __shared__ __align__(16) __bf16 sA2[128 * 132];
  int tid = threadIdx.x, lane = tid & 31, wid = tid >> 5;
  int row0 = blockIdx.x * 128;
  if (tid < 128) { sg[tid] = g[tid]; sbe[tid] = be[tid]; sb1[tid] = b1[tid]; sb2[tid] = b2[tid]; }
  if (tid < 128) {
    int n = row0 + tid;
    float s = 0.f, ss = 0.f;
    if (n < n_tok)
      for (int j = 0; j < 128; ++j) { float v = h[n * 128 + j]; s += v; ss += v * v; }
    float mu = s * (1.f / 128.f);
    smu[tid] = mu;
    srs[tid] = rsqrtf(ss * (1.f / 128.f) - mu * mu + 1e-5f);
  }
  __syncthreads();
  f32x8 acc[8];
#pragma unroll
  for (int t = 0; t < 8; ++t) acc[t] = fzero8();
  for (int kc = 0; kc < 4; ++kc) {            // GEMM1: LN2(h) @ w1
    int kb = kc * 32;
    for (int i = tid; i < 128 * 32; i += 256) {
      int r = i >> 5, kk = i & 31, cc = kb + kk, n = row0 + r;
      float v = (n < n_tok) ? (h[n * 128 + cc] - smu[r]) * srs[r] * sg[cc] + sbe[cc] : 0.f;
      sA[r * 36 + kk] = (__bf16)v;
    }
    for (int i = tid; i < 128 * 32; i += 256) {
      int kk = i >> 7, col = i & 127;
      sW[col * 36 + kk] = (__bf16)w1[(kb + kk) * 128 + col];
    }
    if (kc < 3) prefetch_wchunk(w1, kb + 32, 128, tid);
    else prefetch_wchunk(w2, 0, 128, tid);    // start pulling GEMM2 weights
    __syncthreads();
    bf16x16 a = load_afrag(sA + (wid << 4) * 36, 36, lane);
#pragma unroll
    for (int t = 0; t < 8; ++t)
      acc[t] = wmma_bf16(a, load_bfrag(sW + (t << 4) * 36, 36, lane), acc[t]);
    __syncthreads();
  }
  int hi = lane >> 4, nn = lane & 15;
#pragma unroll
  for (int t = 0; t < 8; ++t) {               // gelu + stage activation for GEMM2
    int col = (t << 4) + nn;
#pragma unroll
    for (int r = 0; r < 8; ++r) {
      int rl = (wid << 4) + (hi << 3) + r;
      sA2[rl * 132 + col] = (__bf16)gelu_f(acc[t][r] + sb1[col]);
    }
  }
  __syncthreads();
#pragma unroll
  for (int t = 0; t < 8; ++t) acc[t] = fzero8();
  for (int kc = 0; kc < 4; ++kc) {            // GEMM2: act @ w2
    int kb = kc * 32;
    for (int i = tid; i < 128 * 32; i += 256) {
      int kk = i >> 7, col = i & 127;
      sW[col * 36 + kk] = (__bf16)w2[(kb + kk) * 128 + col];
    }
    if (kc < 3) prefetch_wchunk(w2, kb + 32, 128, tid);
    __syncthreads();
    bf16x16 a = load_afrag(sA2 + (wid << 4) * 132 + kb, 132, lane);
#pragma unroll
    for (int t = 0; t < 8; ++t)
      acc[t] = wmma_bf16(a, load_bfrag(sW + (t << 4) * 36, 36, lane), acc[t]);
    __syncthreads();
  }
#pragma unroll
  for (int t = 0; t < 8; ++t) {
    int col = (t << 4) + nn;
#pragma unroll
    for (int r = 0; r < 8; ++r) {
      int row = row0 + (wid << 4) + (hi << 3) + r;
      if (row < n_tok) h[row * 128 + col] += acc[t][r] + sb2[col];
    }
  }
}

// =====================  final: out = LN(h) @ lin.w + lin.b  ====================
__global__ void k_final(const float* h, const float* g, const float* be,
                        const float* w, const float* b, float* out, int n_tok) {
  __shared__ float sw[128 * 4];
  __shared__ float sg[128], sbe[128], sb[4];
  int tid = threadIdx.x;
  for (int i = tid; i < 512; i += 256) sw[i] = w[i];
  if (tid < 128) { sg[tid] = g[tid]; sbe[tid] = be[tid]; }
  if (tid < 4) sb[tid] = b[tid];
  __syncthreads();
  int n = blockIdx.x * 256 + tid;
  if (n >= n_tok) return;
  float s = 0.f, ss = 0.f;
  for (int j = 0; j < 128; ++j) { float v = h[n * 128 + j]; s += v; ss += v * v; }
  float mu = s * (1.f / 128.f);
  float rs = rsqrtf(ss * (1.f / 128.f) - mu * mu + 1e-5f);
  float o[4] = {sb[0], sb[1], sb[2], sb[3]};
  for (int j = 0; j < 128; ++j) {
    float v = (h[n * 128 + j] - mu) * rs * sg[j] + sbe[j];
#pragma unroll
    for (int e = 0; e < 4; ++e) o[e] += v * sw[j * 4 + e];
  }
#pragma unroll
  for (int e = 0; e < 4; ++e)
    __builtin_nontemporal_store(o[e], &out[n * 4 + e]);   // stream-out, keep L2 for h
}

// ===============================  host driver  =================================
// Leaf order assumed (setup_inputs insertion order, depth-first):
// 0:x 1:t 2:dt | 3..6 x_emb(fc1.w,fc1.b,fc2.w,fc2.b) | 7..10 t_emb | 11..14 d_emb
// blocks l=0..4 at base 15+16*l:
//   +0 ln1.g +1 ln1.b +2 ln2.g +3 ln2.b +4 kv.w +5 kv.b +6 q.w +7 q.b
//   +8 alpha +9 qkv +10 out.w +11 out.b +12 fc1.w +13 fc1.b +14 fc2.w +15 fc2.b
// 95..98: final ln.g, ln.b, lin.w, lin.b
extern "C" void kernel_launch(void* const* d_in, const int* in_sizes, int n_in,
                              void* d_out, int out_size, void* d_ws, size_t ws_size,
                              hipStream_t stream) {
  const float* const* in = (const float* const*)d_in;
  int N = in_sizes[0] / 8;
  int T = (N + 127) / 128;     // 128-token GEMM tiles
  int B2 = (N + 255) / 256;    // 256-token blocks

  char* p = (char*)d_ws;
  float*  h    = (float*)p;   p += (size_t)N * 128 * sizeof(float);
  __bf16* kv   = (__bf16*)p;  p += (size_t)N * 256 * sizeof(__bf16);
  float*  gpart= (float*)p;   p += (size_t)B2 * 4352 * sizeof(float);
  float*  c    = (float*)p;   p += 512;
  float*  xq   = (float*)p;   p += 4096 * sizeof(float);
  float*  gacc = (float*)p;   p += 4352 * sizeof(float);
  float*  ot   = (float*)p;   p += 4096 * sizeof(float);
  (void)ws_size; (void)n_in; (void)out_size;

  k_timeemb<<<1, 256, 0, stream>>>(in[1], in[2], in[7], in[8], in[9], in[10],
                                   in[11], in[12], in[13], in[14], c);
  k_x_emb<<<T, 256, 0, stream>>>(in[0], in[3], in[4], in[5], in[6], h, N);

  for (int l = 0; l < 5; ++l) {
    int b = 15 + 16 * l;
    k_xq<<<16, 256, 0, stream>>>(c, in[b + 6], in[b + 7], xq);
    k_kv<<<T, 256, 0, stream>>>(h, in[b + 0], in[b + 1], in[b + 4], in[b + 5], kv, N);
    k_slice1<<<B2, 256, 0, stream>>>(kv, xq, in[b + 8], gpart, N);
    k_reduce<<<17, 256, 0, stream>>>(gpart, gacc, B2);
    k_slice_mix<<<1, 256, 0, stream>>>(gacc, in[b + 9], ot);
    k_attn_out<<<T, 256, 0, stream>>>(h, kv, xq, in[b + 8], ot, in[b + 10],
                                      in[b + 11], N);
    k_mlp<<<T, 256, 0, stream>>>(h, in[b + 2], in[b + 3], in[b + 12], in[b + 13],
                                 in[b + 14], in[b + 15], N);
  }
  k_final<<<B2, 256, 0, stream>>>(h, in[95], in[96], in[97], in[98],
                                  (float*)d_out, N);
}
